// MixtureOfExperts_34059090657979
// MI455X (gfx1250) — matrix operand order repaired and checked by reading, not compiled, not run on hardware
//
#include <hip/hip_runtime.h>
#include <hip/hip_bf16.h>
#include <math.h>

// ---------------------------------------------------------------------------
// MoE (top-2, E=8, D=1024, F=4096, T=8192) for MI455X / gfx1250.
//   1. x -> bf16; w1 -> [E][F][D] bf16, w2 -> [E][D][F] bf16 (pre-transposed
//      so WMMA A/B fragments are contiguous 16B per-lane loads). 128 MB of
//      bf16 weights stay resident in MI455X's 192 MB L2.
//   2. Gate: per-token top-2 softmax, atomic append to per-expert lists.
//   3. Fused expert kernel per (expert, 64-token tile):
//      - gathered X tile (64x1024 bf16) staged into LDS via
//        global_load_async_to_lds_b128 (ASYNCcnt) once per block,
//      - per 256-wide F chunk: H = gelu(X@W1+b1) via v_wmma_f32_16x16x32_bf16
//        (A from LDS, B from L2), H -> LDS bf16,
//      - OUT[64,1024] accumulated in VGPRs (32 C-tiles/wave),
//      - combined into d_out with f32 atomics scaled by gate weight.
// Workspace: ~152 MB. Dynamic LDS: 168 KB/workgroup (CDNA5 WGP has 320 KB).
// ---------------------------------------------------------------------------

typedef __attribute__((ext_vector_type(16))) __bf16 v16bf;
typedef __attribute__((ext_vector_type(8)))  float  v8f;

#define T_TOK 8192
#define DIM   1024
#define FDIM  4096
#define NEXP  8
#define MTILE 64
#define FCH   256
#define XP    1048  // LDS X row stride (elems): 2096B, 16B aligned, conflict-free
#define HP    264   // LDS H row stride (elems): 528B, 16B aligned, conflict-free

#define XELE  (MTILE * XP)              // X tile elems in LDS
#define LDSBYTES ((MTILE * XP + MTILE * HP) * 2)

union AF { v16bf v; uint4 q[2]; };

// ---------------------------------------------------------------------------
__global__ __launch_bounds__(256) void cvt_x_kernel(const float* __restrict__ x,
                                                    __bf16* __restrict__ xb) {
  int i = (blockIdx.x * 256 + threadIdx.x) * 4;
  float4 v = *(const float4*)(x + i);
  union { __bf16 b[4]; uint2 u; } p;
  p.b[0] = (__bf16)v.x; p.b[1] = (__bf16)v.y;
  p.b[2] = (__bf16)v.z; p.b[3] = (__bf16)v.w;
  *(uint2*)(xb + i) = p.u;
}

// ---------------------------------------------------------------------------
// Per-expert transpose + fp32->bf16: in [z][R][C] -> out [z][C][R]
__global__ __launch_bounds__(256) void transpose_bf16_kernel(
    const float* __restrict__ in, __bf16* __restrict__ outp, int R, int C) {
  __shared__ float tile[32][33];
  int z = blockIdx.z;
  const float* ine  = in   + (size_t)z * R * C;
  __bf16*      oute = outp + (size_t)z * R * C;
  int c0 = blockIdx.x * 32, r0 = blockIdx.y * 32;
  int tx = threadIdx.x & 31, ty = threadIdx.x >> 5;
  #pragma unroll
  for (int i = 0; i < 32; i += 8)
    tile[ty + i][tx] = ine[(size_t)(r0 + ty + i) * C + c0 + tx];
  __syncthreads();
  #pragma unroll
  for (int i = 0; i < 32; i += 8)
    oute[(size_t)(c0 + ty + i) * R + r0 + tx] = (__bf16)tile[tx][ty + i];
}

// ---------------------------------------------------------------------------
__global__ __launch_bounds__(256) void gate_kernel(
    const float* __restrict__ x, const float* __restrict__ gate_w,
    int* __restrict__ ids, float* __restrict__ wts, int* __restrict__ counts) {
  __shared__ float gw[DIM * NEXP];                 // 32 KB
  for (int i = threadIdx.x; i < DIM * NEXP; i += 256) gw[i] = gate_w[i];
  __syncthreads();

  int t = blockIdx.x * 256 + threadIdx.x;
  const float* xr = x + (size_t)t * DIM;
  float lg[NEXP];
  #pragma unroll
  for (int e = 0; e < NEXP; e++) lg[e] = 0.0f;

  for (int d = 0; d < DIM; d += 4) {
    float4 xv = *(const float4*)(xr + d);
    #pragma unroll
    for (int e = 0; e < NEXP; e++) {
      lg[e] += xv.x * gw[(d + 0) * NEXP + e] + xv.y * gw[(d + 1) * NEXP + e]
             + xv.z * gw[(d + 2) * NEXP + e] + xv.w * gw[(d + 3) * NEXP + e];
    }
  }
  int i0 = 0;
  #pragma unroll
  for (int e = 1; e < NEXP; e++) if (lg[e] > lg[i0]) i0 = e;
  int i1 = (i0 == 0) ? 1 : 0;
  #pragma unroll
  for (int e = 0; e < NEXP; e++) if (e != i0 && lg[e] > lg[i1]) i1 = e;

  float w0 = 1.0f / (1.0f + __expf(lg[i1] - lg[i0]));
  float w1 = 1.0f - w0;

  int s0 = atomicAdd(&counts[i0], 1);
  ids[i0 * T_TOK + s0] = t;  wts[i0 * T_TOK + s0] = w0;
  int s1 = atomicAdd(&counts[i1], 1);
  ids[i1 * T_TOK + s1] = t;  wts[i1 * T_TOK + s1] = w1;
}

// ---------------------------------------------------------------------------
// Fused expert FFN. Grid: E * (T/MTILE) blocks, 256 threads (8 waves).
// Wave w: row group rgrp=w>>1 (16 rows), column half cgrp=w&1.
__global__ __launch_bounds__(256) void moe_expert_kernel(
    const __bf16* __restrict__ xb,  const __bf16* __restrict__ w1t,
    const __bf16* __restrict__ w2t, const float* __restrict__ b1,
    const float* __restrict__ b2,   const int* __restrict__ ids,
    const float* __restrict__ wts,  const int* __restrict__ counts,
    float* __restrict__ out) {
  const int MAXT = T_TOK / MTILE;
  int e    = blockIdx.x / MAXT;
  int tile = blockIdx.x % MAXT;
  int cnt  = counts[e];
  if (tile * MTILE >= cnt) return;

  extern __shared__ __bf16 dynlds[];       // [MTILE*XP] X | [MTILE*HP] H
  __bf16* Xlds = dynlds;
  __bf16* Hlds = dynlds + XELE;
  __shared__ int    sid[MTILE];
  __shared__ float  swt[MTILE];

  int tid  = threadIdx.x;
  int lane = tid & 31;
  int w    = tid >> 5;
  int rgrp = w >> 1;        // 0..3
  int cgrp = w & 1;         // 0..1

  if (tid < MTILE) {
    int slot = tile * MTILE + tid;
    if (slot < cnt) { sid[tid] = ids[e * T_TOK + slot]; swt[tid] = wts[e * T_TOK + slot]; }
    else            { sid[tid] = ids[e * T_TOK];        swt[tid] = 0.0f; }
  }
  __syncthreads();

  // ---- stage gathered X tile into LDS with CDNA5 async loads (ASYNCcnt) ---
  {
    int row  = tid >> 2;                       // 4 threads per row
    int quad = tid & 3;                        // 256 elems each
    const __bf16* xrow = xb + (size_t)sid[row] * DIM + quad * 256;
    unsigned ldsbase = (unsigned)(size_t)(&Xlds[row * XP + quad * 256]);
    #pragma unroll 8
    for (int i = 0; i < 32; i++) {            // 32 x 16B = 512B per thread
      const __bf16* g = xrow + i * 8;
      unsigned l = ldsbase + i * 16;
      asm volatile("global_load_async_to_lds_b128 %0, %1, off"
                   :: "v"(l), "v"(g) : "memory");
    }
    asm volatile("s_wait_asynccnt 0x0" ::: "memory");
  }
  __syncthreads();

  // A-fragment addressing (16-bit A layout): lane&15 = row M,
  // lanes 16-31 take K offset +8 within each 32-wide K step.
  int am    = lane & 15;
  int akoff = (lane >> 4) ? 8 : 0;
  // B-fragment addressing (16-bit B layout): lane&15 = col N,
  // lanes 16-31 take K offset +8 within each 32-wide K step.
  int bn    = lane & 15;
  int bkoff = (lane >> 4) ? 8 : 0;

  const __bf16* w1e = w1t + (size_t)e * FDIM * DIM;  // [F][D]
  const __bf16* w2e = w2t + (size_t)e * DIM * FDIM;  // [D][F]
  const float*  b1e = b1 + (size_t)e * FDIM;
  const float*  b2e = b2 + (size_t)e * DIM;

  v8f acc[32];
  #pragma unroll
  for (int t = 0; t < 32; t++)
    #pragma unroll
    for (int j = 0; j < 8; j++) acc[t][j] = 0.0f;

  for (int fc = 0; fc < FDIM; fc += FCH) {
    // light prefetch of next F-chunk weight rows (global_prefetch_b8)
    if (fc + FCH < FDIM) {
      __builtin_prefetch(w1e + (size_t)(fc + FCH + cgrp * 128 + bn) * DIM, 0, 1);
      __builtin_prefetch(w2e + (size_t)(cgrp * 512 + (lane >> 4) * 256 + bn) * FDIM
                             + fc + FCH, 0, 1);
    }
    // ---------------- GEMM1: 16 rows x 128 H-cols per wave ----------------
    v8f c1[8];
    #pragma unroll
    for (int j = 0; j < 8; j++)
      #pragma unroll
      for (int v = 0; v < 8; v++) c1[j][v] = 0.0f;

    for (int kk = 0; kk < DIM; kk += 32) {
      AF a;
      const __bf16* ar = &Xlds[(rgrp * 16 + am) * XP + kk + akoff];
      a.q[0] = *(const uint4*)(ar);
      a.q[1] = *(const uint4*)(ar + 16);
      #pragma unroll
      for (int j = 0; j < 8; j++) {
        const __bf16* brow = w1e + (size_t)(fc + cgrp * 128 + j * 16 + bn) * DIM;
        AF b;
        b.q[0] = *(const uint4*)(brow + kk + bkoff);
        b.q[1] = *(const uint4*)(brow + kk + bkoff + 16);
        c1[j] = __builtin_amdgcn_wmma_f32_16x16x32_bf16(
            false, a.v, false, b.v, (short)0, c1[j], false, false);
      }
    }
    // bias + exact GELU + bf16, store H tile to LDS
    #pragma unroll
    for (int j = 0; j < 8; j++) {
      int col = cgrp * 128 + j * 16 + (lane & 15);
      float bias = b1e[fc + col];
      #pragma unroll
      for (int v = 0; v < 8; v++) {
        int m = v + ((lane >> 4) << 3);          // C layout: lanes>=16 -> M+8
        float hv = c1[j][v] + bias;
        hv = 0.5f * hv * (1.0f + erff(hv * 0.70710678118654752f));
        Hlds[(rgrp * 16 + m) * HP + col] = (__bf16)hv;
      }
    }
    __syncthreads();

    // --------- GEMM2: acc[16 x 512] += H[16,256] @ W2[256,512] ------------
    AF ah[8];
    #pragma unroll
    for (int ks = 0; ks < 8; ks++) {
      const __bf16* hrow = &Hlds[(rgrp * 16 + am) * HP + ks * 32 + akoff];
      ah[ks].q[0] = *(const uint4*)(hrow);
      ah[ks].q[1] = *(const uint4*)(hrow + 16);
    }
    #pragma unroll
    for (int t = 0; t < 32; t++) {
      const __bf16* brow = w2e + (size_t)(cgrp * 512 + t * 16 + bn) * FDIM + fc;
      #pragma unroll
      for (int ks = 0; ks < 8; ks++) {
        AF b;
        b.q[0] = *(const uint4*)(brow + ks * 32 + bkoff);
        b.q[1] = *(const uint4*)(brow + ks * 32 + bkoff + 16);
        acc[t] = __builtin_amdgcn_wmma_f32_16x16x32_bf16(
            false, ah[ks].v, false, b.v, (short)0, acc[t], false, false);
      }
    }
    __syncthreads();
  }

  // ---------------- writeback: out[tok] += wt * (acc + b2) ----------------
  #pragma unroll
  for (int t = 0; t < 32; t++) {
    int col = cgrp * 512 + t * 16 + (lane & 15);
    float bias = b2e[col];
    #pragma unroll
    for (int v = 0; v < 8; v++) {
      int m   = v + ((lane >> 4) << 3);
      int row = rgrp * 16 + m;
      float val = swt[row] * (acc[t][v] + bias);
      atomicAdd(&out[(size_t)sid[row] * DIM + col], val);
    }
  }
}

// ---------------------------------------------------------------------------
extern "C" void kernel_launch(void* const* d_in, const int* in_sizes, int n_in,
                              void* d_out, int out_size, void* d_ws, size_t ws_size,
                              hipStream_t stream) {
  const float* x      = (const float*)d_in[0];
  const float* gate_w = (const float*)d_in[1];
  const float* w1     = (const float*)d_in[2];
  const float* b1     = (const float*)d_in[3];
  const float* w2     = (const float*)d_in[4];
  const float* b2     = (const float*)d_in[5];
  float* out = (float*)d_out;

  char* ws = (char*)d_ws;
  const size_t SZ_XB = (size_t)T_TOK * DIM * 2;          // 16 MB
  const size_t SZ_W  = (size_t)NEXP * DIM * FDIM * 2;    // 64 MB each
  const size_t SZ_L  = (size_t)NEXP * T_TOK * 4;         // 256 KB each
  __bf16* xb   = (__bf16*)(ws);
  __bf16* w1t  = (__bf16*)(ws + SZ_XB);
  __bf16* w2t  = (__bf16*)(ws + SZ_XB + SZ_W);
  int*    ids  = (int*)  (ws + SZ_XB + 2 * SZ_W);
  float*  wts  = (float*)(ws + SZ_XB + 2 * SZ_W + SZ_L);
  int*  counts = (int*)  (ws + SZ_XB + 2 * SZ_W + 2 * SZ_L);

  hipMemsetAsync(counts, 0, NEXP * sizeof(int), stream);
  hipMemsetAsync(out, 0, (size_t)T_TOK * DIM * sizeof(float), stream);

  cvt_x_kernel<<<(T_TOK * DIM) / (256 * 4), 256, 0, stream>>>(x, xb);
  transpose_bf16_kernel<<<dim3(FDIM / 32, DIM / 32, NEXP), 256, 0, stream>>>(w1, w1t, DIM, FDIM);
  transpose_bf16_kernel<<<dim3(DIM / 32, FDIM / 32, NEXP), 256, 0, stream>>>(w2, w2t, FDIM, DIM);
  gate_kernel<<<T_TOK / 256, 256, 0, stream>>>(x, gate_w, ids, wts, counts);
  moe_expert_kernel<<<NEXP * (T_TOK / MTILE), 256, LDSBYTES, stream>>>(
      xb, w1t, w2t, b1, b2, ids, wts, counts, out);
}
